// RoutedDimPool_11158325035228
// MI455X (gfx1250) — matrix operand order repaired and checked by reading, not compiled, not run on hardware
//
#include <hip/hip_runtime.h>
#include <hip/hip_bf16.h>

typedef __attribute__((ext_vector_type(8)))  __bf16 v8bf;
typedef __attribute__((ext_vector_type(16))) __bf16 v16bf;
typedef __attribute__((ext_vector_type(8)))  float  v8f;

#define NTHREADS 512
#define WAVES    16

// ---------------------------------------------------------------------------
// WMMA helpers (CDNA5 gfx1250, wave32).
// A 16x32 bf16 fragment: lane l holds row m=l%16; elems e -> K = g*16+e,
// g = l/16 (same k order as B).
// B 32x16 bf16 fragment: lane l holds col n=l%16; elems e -> K = g*16+e.
// C/D f32: lane l holds col n=l%16, rows m = g*8 + r (r = vgpr index).
// ---------------------------------------------------------------------------
__device__ __forceinline__ v8f wmma_bf16(v16bf a, v16bf b, v8f c) {
  return __builtin_amdgcn_wmma_f32_16x16x32_bf16(false, a, false, b,
                                                 (short)0, c, false, false);
}

__device__ __forceinline__ v16bf afrag(const __bf16* row, int kbase, int g) {
  v8bf lo = *(const v8bf*)(row + kbase + g * 16);
  v8bf hi = *(const v8bf*)(row + kbase + g * 16 + 8);
  return __builtin_shufflevector(lo, hi, 0,1,2,3,4,5,6,7,8,9,10,11,12,13,14,15);
}

__device__ __forceinline__ v16bf bfrag_row(const __bf16* row, int kbase, int g) {
  v8bf lo = *(const v8bf*)(row + kbase + g * 16);
  v8bf hi = *(const v8bf*)(row + kbase + g * 16 + 8);
  return __builtin_shufflevector(lo, hi, 0,1,2,3,4,5,6,7,8,9,10,11,12,13,14,15);
}

__device__ __forceinline__ float wred(float v) {
#pragma unroll
  for (int o = 16; o > 0; o >>= 1) v += __shfl_xor(v, o, 32);
  return v;
}

__device__ __forceinline__ v8f vzero8() {
  v8f z = {0.f,0.f,0.f,0.f,0.f,0.f,0.f,0.f};
  return z;
}

// Async global->LDS copy of 16B per lane (CDNA5 GLOBAL_LOAD_ASYNC_TO_LDS_B128,
// tracked by ASYNCcnt). ldsOff is the LDS byte offset (low 32 bits of a
// generic shared pointer), gaddr the 64-bit global address.
__device__ __forceinline__ void async_copy_b128(unsigned ldsOff,
                                                const void* gaddr) {
  unsigned long long ga = (unsigned long long)(size_t)gaddr;
  asm volatile("global_load_async_to_lds_b128 %0, %1, off"
               :: "v"(ldsOff), "v"(ga)
               : "memory");
}
__device__ __forceinline__ void async_wait0() {
  asm volatile("s_wait_asynccnt 0x0" ::: "memory");
}

// ---------------------------------------------------------------------------
// Weight packer: f32 row-major W[K][N] -> bf16 WMMA-B-fragment order.
// Packed element index = ((nt*KT + kt)*32 + lane)*16 + e
//   holds W[kt*32 + (lane/16)*16 + e][nt*16 + lane%16]
// so each lane's B fragment is one contiguous 32-byte read.
// ---------------------------------------------------------------------------
__global__ void pack_weight_kernel(const float* __restrict__ W,
                                   __bf16* __restrict__ out, int K, int N) {
  long idx = (long)blockIdx.x * blockDim.x + threadIdx.x;
  long total = (long)K * N;
  if (idx >= total) return;
  int e = (int)(idx & 15);
  long t = idx >> 4;
  int lane = (int)(t & 31);
  t >>= 5;
  int KT = K >> 5;
  int kt = (int)(t % KT);
  int nt = (int)(t / KT);
  int g = lane >> 4;
  int k = kt * 32 + g * 16 + e;
  int n = nt * 16 + (lane & 15);
  out[idx] = (__bf16)W[(long)k * N + n];
}

// ---------------------------------------------------------------------------
// LayerNorm f32 LDS -> bf16 LDS (one wave per 2 rows, 16 waves = 32 tokens).
// ---------------------------------------------------------------------------
__device__ void layernorm_bf16(const float* __restrict__ xres,
                               __bf16* __restrict__ xb,
                               const float* __restrict__ w,
                               const float* __restrict__ b,
                               int wave, int lane) {
  for (int rr = 0; rr < 2; ++rr) {
    int row = wave * 2 + rr;
    const float* xr = xres + row * 1024;
    float s = 0.f;
    for (int d = lane; d < 1024; d += 32) s += xr[d];
    s = wred(s);
    float mean = s * (1.0f / 1024.0f);
    float v = 0.f;
    for (int d = lane; d < 1024; d += 32) {
      float t = xr[d] - mean;
      v += t * t;
    }
    v = wred(v);
    float rstd = rsqrtf(v * (1.0f / 1024.0f) + 1e-5f);
    for (int d = lane; d < 1024; d += 32)
      xb[row * 1024 + d] = (__bf16)((xr[d] - mean) * rstd * w[d] + b[d]);
  }
}

// ---------------------------------------------------------------------------
// Fused router: 2 transformer layers + logits/argmax + masks + compression.
// One block per sequence (1024 blocks), 512 threads = 16 waves.
// Dynamic LDS layout (bytes):
//   0      : xres  f32 [32][1024]            131072
//   131072 : xb    bf16[32][1024]             65536
//   196608 : region (attn buffers | hbuf)    102400
//     +0     qb   bf16[32][128]   (2 heads)    8192
//     +8192  kb   bf16[32][128]                8192
//     +16384 vT   bf16[128][32]                8192
//     +24576 sbuf f32 [2][32][32]              8192
//     +32768 pbuf bf16[2][32][32]              4096
//     +36864 obuf bf16[32][1024]              65536
//     (hbuf bf16[32][1024] aliases +0)
//   299008 : maskl f32[32], 299136: chos i32[32], 299264: misc
// total 299328 B  (< 320KB WGP LDS)
// ---------------------------------------------------------------------------
#define RTR_LDS 299328

__global__ void __launch_bounds__(NTHREADS)
router_kernel(const float* __restrict__ x, const float* __restrict__ amask,
              const float* __restrict__ ln1w, const float* __restrict__ ln1b,
              const float* __restrict__ bqkv, const float* __restrict__ bo,
              const float* __restrict__ ln2w, const float* __restrict__ ln2b,
              const float* __restrict__ b1,   const float* __restrict__ b2,
              const float* __restrict__ rfw,  const float* __restrict__ rfb,
              const __bf16* __restrict__ pk_qkv, const __bf16* __restrict__ pk_wo,
              const __bf16* __restrict__ pk_w1,  const __bf16* __restrict__ pk_w2,
              int* __restrict__ chosen_g, float* __restrict__ masks_out,
              float* __restrict__ comp_out) {
  extern __shared__ char smem_raw[];
  float*  xres  = (float*)smem_raw;
  __bf16* xb    = (__bf16*)(smem_raw + 131072);
  char*   region = smem_raw + 196608;
  __bf16* qb    = (__bf16*)region;
  __bf16* kb    = (__bf16*)(region + 8192);
  __bf16* vT    = (__bf16*)(region + 16384);
  float*  sbuf  = (float*)(region + 24576);
  __bf16* pbuf  = (__bf16*)(region + 32768);
  __bf16* obuf  = (__bf16*)(region + 36864);
  __bf16* hbuf  = (__bf16*)region;
  float*  maskl = (float*)(smem_raw + 299008);
  int*    chos  = (int*)(smem_raw + 299136);
  float*  miscf = (float*)(smem_raw + 299264);

  const int seq  = blockIdx.x;
  const int tid  = threadIdx.x;
  const int wave = tid >> 5;
  const int lane = tid & 31;
  const int g    = lane >> 4;
  const int n    = lane & 15;
  const int m0   = g * 8;

  // ---- async DMA of the residual stream into LDS (128KB, 16B/lane chunks)
  const float* xg = x + (size_t)seq * 32 * 1024;
  {
    unsigned ldsBase = (unsigned)(size_t)(void*)xres;
    for (int i = tid; i < 8192; i += NTHREADS)
      async_copy_b128(ldsBase + (unsigned)i * 16u, xg + (size_t)i * 4);
    async_wait0();
  }
  if (tid < 32) maskl[tid] = amask[seq * 32 + tid];
  __syncthreads();

  for (int l = 0; l < 2; ++l) {
    // ---- LN1 -> xb (bf16) ----
    layernorm_bf16(xres, xb, ln1w + l * 1024, ln1b + l * 1024, wave, lane);
    __syncthreads();

    const v16bf* pqkv = (const v16bf*)pk_qkv + (size_t)l * (1024 * 3072 / 16);
    const __bf16* xrow0 = xb + n * 1024;
    const __bf16* xrow1 = xb + (16 + n) * 1024;

    // ---- attention, 2 heads per phase ----
    for (int ph = 0; ph < 8; ++ph) {
      const int h0 = ph * 2;
      // qkv partial GEMM: 24 col-tiles for these 2 heads; both row-tiles per
      // ct so each B fragment is loaded once and feeds two WMMAs.
      for (int ct = wave; ct < 24; ct += WAVES) {
        int grp = ct >> 3, sub = ct & 7;
        int gnt = grp * 64 + h0 * 4 + sub;   // global col-tile in [0,192)
        v8f acc0 = vzero8(), acc1 = vzero8();
        for (int kt = 0; kt < 32; ++kt) {
          v16bf b  = pqkv[((size_t)(gnt * 32 + kt)) * 32 + lane];
          v16bf a0 = afrag(xrow0, kt * 32, g);
          v16bf a1 = afrag(xrow1, kt * 32, g);
          acc0 = wmma_bf16(a0, b, acc0);
          acc1 = wmma_bf16(a1, b, acc1);
        }
        int gcol = gnt * 16 + n;
        float bias = bqkv[l * 3072 + gcol];
        int c = sub * 16 + n;
        if (grp == 0) {
#pragma unroll
          for (int r = 0; r < 8; ++r) {
            qb[(m0 + r) * 128 + c]      = (__bf16)((acc0[r] + bias) * 0.125f);
            qb[(16 + m0 + r) * 128 + c] = (__bf16)((acc1[r] + bias) * 0.125f);
          }
        } else if (grp == 1) {
#pragma unroll
          for (int r = 0; r < 8; ++r) {
            kb[(m0 + r) * 128 + c]      = (__bf16)(acc0[r] + bias);
            kb[(16 + m0 + r) * 128 + c] = (__bf16)(acc1[r] + bias);
          }
        } else {
          v8bf pv0, pv1;
#pragma unroll
          for (int r = 0; r < 8; ++r) {
            pv0[r] = (__bf16)(acc0[r] + bias);
            pv1[r] = (__bf16)(acc1[r] + bias);
          }
          *(v8bf*)(vT + c * 32 + m0)      = pv0;   // transposed V
          *(v8bf*)(vT + c * 32 + 16 + m0) = pv1;
        }
      }
      __syncthreads();

      // scores s = q k^T (+mask bias): 2 heads x 2x2 tiles -> waves 0..7
      if (wave < 8) {
        int hh = wave >> 2, rt = (wave >> 1) & 1, ct = wave & 1;
        const __bf16* arow = qb + (rt * 16 + n) * 128;
        v8f acc = vzero8();
#pragma unroll
        for (int kt = 0; kt < 2; ++kt) {
          v16bf a = afrag(arow, hh * 64 + kt * 32, g);
          v16bf b = bfrag_row(kb + (ct * 16 + n) * 128, hh * 64 + kt * 32, g);
          acc = wmma_bf16(a, b, acc);
        }
        int col = ct * 16 + n;
        float sb = (maskl[col] > 0.f) ? 0.f : -1e9f;
#pragma unroll
        for (int r = 0; r < 8; ++r)
          sbuf[hh * 1024 + (rt * 16 + m0 + r) * 32 + col] = acc[r] + sb;
      }
      __syncthreads();

      // softmax rows (64 rows)
      if (tid < 64) {
        float* srow = sbuf + (tid >> 5) * 1024 + (tid & 31) * 32;
        __bf16* prow = pbuf + (tid >> 5) * 1024 + (tid & 31) * 32;
        float mx = srow[0];
        for (int jj = 1; jj < 32; ++jj) mx = fmaxf(mx, srow[jj]);
        float sm = 0.f;
        for (int jj = 0; jj < 32; ++jj) sm += __expf(srow[jj] - mx);
        float inv = 1.0f / sm;
        for (int jj = 0; jj < 32; ++jj)
          prow[jj] = (__bf16)(__expf(srow[jj] - mx) * inv);
      }
      __syncthreads();

      // o = p @ v : 2 heads x 2x4 tiles -> all 16 waves, K=32 (one WMMA)
      {
        int hh = wave >> 3, rt = (wave >> 2) & 1, ct = wave & 3;
        const __bf16* arow = pbuf + hh * 1024 + (rt * 16 + n) * 32;
        v16bf a = afrag(arow, 0, g);
        v16bf b = bfrag_row(vT + (hh * 64 + ct * 16 + n) * 32, 0, g);
        v8f acc = vzero8();
        acc = wmma_bf16(a, b, acc);
        int col = (h0 + hh) * 64 + ct * 16 + n;
#pragma unroll
        for (int r = 0; r < 8; ++r)
          obuf[(rt * 16 + m0 + r) * 1024 + col] = (__bf16)acc[r];
      }
      __syncthreads();
    }

    // ---- output projection + residual: xres += obuf @ wo + bo ----
    const v16bf* pwo = (const v16bf*)pk_wo + (size_t)l * (1024 * 1024 / 16);
    const __bf16* orow0 = obuf + n * 1024;
    const __bf16* orow1 = obuf + (16 + n) * 1024;
    for (int jc = 0; jc < 4; ++jc) {
      int ct = wave + jc * WAVES;    // 0..63
      v8f acc0 = vzero8(), acc1 = vzero8();
      for (int kt = 0; kt < 32; ++kt) {
        v16bf b  = pwo[((size_t)(ct * 32 + kt)) * 32 + lane];
        v16bf a0 = afrag(orow0, kt * 32, g);
        v16bf a1 = afrag(orow1, kt * 32, g);
        acc0 = wmma_bf16(a0, b, acc0);
        acc1 = wmma_bf16(a1, b, acc1);
      }
      int col = ct * 16 + n;
      float bias = bo[l * 1024 + col];
#pragma unroll
      for (int r = 0; r < 8; ++r) {
        xres[(m0 + r) * 1024 + col]      += acc0[r] + bias;
        xres[(16 + m0 + r) * 1024 + col] += acc1[r] + bias;
      }
    }
    __syncthreads();

    // ---- LN2 -> xb ----
    layernorm_bf16(xres, xb, ln2w + l * 1024, ln2b + l * 1024, wave, lane);
    __syncthreads();

    // ---- MLP: xres += gelu(xb@w1+b1)@w2 + b2, K-chunked (4 x 1024).
    // Output accumulates straight into the f32 residual in LDS (each
    // (row,col) is owned by one fixed lane) -> no persistent VGPR acc.
    const v16bf* pw1 = (const v16bf*)pk_w1 + (size_t)l * (1024 * 4096 / 16);
    const v16bf* pw2 = (const v16bf*)pk_w2 + (size_t)l * (4096 * 1024 / 16);

    for (int kc = 0; kc < 4; ++kc) {
      // H chunk = gelu(xb @ w1[:, kc*1024:+1024] + b1)
      for (int jc = 0; jc < 4; ++jc) {
        int ct = wave + jc * WAVES;
        v8f acc0 = vzero8(), acc1 = vzero8();
        for (int kt = 0; kt < 32; ++kt) {
          v16bf b  = pw1[((size_t)((kc * 64 + ct) * 32 + kt)) * 32 + lane];
          v16bf a0 = afrag(xrow0, kt * 32, g);
          v16bf a1 = afrag(xrow1, kt * 32, g);
          acc0 = wmma_bf16(a0, b, acc0);
          acc1 = wmma_bf16(a1, b, acc1);
        }
        int col = ct * 16 + n;
        float bias = b1[l * 4096 + kc * 1024 + col];
#pragma unroll
        for (int r = 0; r < 8; ++r) {
          float h0v = acc0[r] + bias;
          float h1v = acc1[r] + bias;
          float g0 = 0.5f * h0v * (1.0f + tanhf(0.7978845608f * (h0v + 0.044715f * h0v * h0v * h0v)));
          float g1 = 0.5f * h1v * (1.0f + tanhf(0.7978845608f * (h1v + 0.044715f * h1v * h1v * h1v)));
          hbuf[(m0 + r) * 1024 + col]      = (__bf16)g0;
          hbuf[(16 + m0 + r) * 1024 + col] = (__bf16)g1;
        }
      }
      __syncthreads();
      // xres += H @ w2[kc*1024:+1024, :]  (+ b2 once, on first chunk)
      const __bf16* hrow0 = hbuf + n * 1024;
      const __bf16* hrow1 = hbuf + (16 + n) * 1024;
      for (int jc = 0; jc < 4; ++jc) {
        int ct = wave + jc * WAVES;
        v8f acc0 = vzero8(), acc1 = vzero8();
        for (int kt = 0; kt < 32; ++kt) {
          v16bf b  = pw2[((size_t)(ct * 128 + kc * 32 + kt)) * 32 + lane];
          v16bf a0 = afrag(hrow0, kt * 32, g);
          v16bf a1 = afrag(hrow1, kt * 32, g);
          acc0 = wmma_bf16(a0, b, acc0);
          acc1 = wmma_bf16(a1, b, acc1);
        }
        int col = ct * 16 + n;
        float bias = (kc == 0) ? b2[l * 1024 + col] : 0.f;
#pragma unroll
        for (int r = 0; r < 8; ++r) {
          xres[(m0 + r) * 1024 + col]      += acc0[r] + bias;
          xres[(16 + m0 + r) * 1024 + col] += acc1[r] + bias;
        }
      }
      __syncthreads();
    }
  } // layers

  // ---- logits / argmax (N=3: plain VALU dot + wave reduce) ----
  for (int jj = 0; jj < 2; ++jj) {
    int t = wave + jj * WAVES;
    const float* xr = xres + t * 1024;
    float s0 = 0.f, s1 = 0.f, s2 = 0.f;
    for (int d = lane; d < 1024; d += 32) {
      float xv = xr[d];
      s0 += xv * rfw[d * 3 + 0];
      s1 += xv * rfw[d * 3 + 1];
      s2 += xv * rfw[d * 3 + 2];
    }
    s0 = wred(s0); s1 = wred(s1); s2 = wred(s2);
    if (lane == 0) {
      s0 += rfb[0]; s1 += rfb[1]; s2 += rfb[2];
      int c = 0; float best = s0;
      if (s1 > best) { best = s1; c = 1; }
      if (s2 > best) { c = 2; }
      chos[t] = c;
      chosen_g[seq * 32 + t] = c;
    }
  }
  __syncthreads();

  if (tid == 0) {
    float ms = 0.f, cs = 0.f;
    for (int t = 0; t < 32; ++t) {
      ms += maskl[t];
      cs += ((float)chos[t] * (1.0f / 3.0f)) * maskl[t];
    }
    miscf[0] = ms;
    comp_out[seq] = cs / ms;
  }
  __syncthreads();
  if (tid < 32) {
    float ms = miscf[0];
    int c = chos[tid];
    float mo;
    if (c == 0)      mo = maskl[tid];
    else if (c == 1) mo = ((float)tid < ms * 0.5f)  ? 1.f : 0.f;
    else             mo = ((float)tid < ms * 0.25f) ? 1.f : 0.f;
    masks_out[seq * 32 + tid] = mo;
  }
}

// ---------------------------------------------------------------------------
// Finalize: WMMA pool2/pool4 GEMMs + routed gather per token.
// Dynamic LDS: xm bf16[32][1024] (65536) | p2 f32[16][1024] (65536)
//              | p4 f32[8][1024] (32768) | maskl(128) | chos(128) | misc
// ---------------------------------------------------------------------------
#define FIN_LDS 164352

__global__ void __launch_bounds__(NTHREADS)
finalize_kernel(const float* __restrict__ x, const float* __restrict__ amask,
                const __bf16* __restrict__ pk_p2, const __bf16* __restrict__ pk_p4,
                const float* __restrict__ p2b, const float* __restrict__ p4b,
                const float* __restrict__ femb, const int* __restrict__ chosen_g,
                float* __restrict__ out) {
  extern __shared__ char smem_raw[];
  __bf16* xm    = (__bf16*)smem_raw;
  float*  p2    = (float*)(smem_raw + 65536);
  float*  p4    = (float*)(smem_raw + 131072);
  float*  maskl = (float*)(smem_raw + 163840);
  int*    chos  = (int*)(smem_raw + 163968);
  float*  miscf = (float*)(smem_raw + 164096);

  const int seq = blockIdx.x;
  const int tid = threadIdx.x;
  const int wave = tid >> 5, lane = tid & 31;
  const int g = lane >> 4, n = lane & 15, m0 = g * 8;
  const float* xg = x + (size_t)seq * 32768;

  if (tid < 32) {
    maskl[tid] = amask[seq * 32 + tid];
    chos[tid] = chosen_g[seq * 32 + tid];
  }
  if (tid == 0) {
    float ms = 0.f;
    for (int t = 0; t < 32; ++t) ms += amask[seq * 32 + t];
    miscf[0] = ms;
  }
  __syncthreads();

  for (int i = tid; i < 32768; i += NTHREADS) {
    int t = i >> 10;
    xm[i] = (__bf16)(xg[i] * maskl[t]);
  }
  __syncthreads();

  const v16bf* pp2 = (const v16bf*)pk_p2;   // K=2048 (KT=64), N=1024 (NT=64)
  const v16bf* pp4 = (const v16bf*)pk_p4;   // K=4096 (KT=128), N=1024

#pragma unroll
  for (int j = 0; j < 8; ++j) {
    int t = wave + j * WAVES;
    if (t < 64) {
      int ct = t;
      v8f acc = vzero8();
      for (int kt = 0; kt < 64; ++kt) {
        int tok = 2 * n + (kt >> 5);                 // reshape (16, 2*1024)
        const __bf16* arow = xm + tok * 1024 + (kt & 31) * 32;
        v16bf a = afrag(arow, 0, g);
        v16bf b = pp2[((size_t)(ct * 64 + kt)) * 32 + lane];
        acc = wmma_bf16(a, b, acc);
      }
      int col = ct * 16 + n;
      float bias = p2b[col];
#pragma unroll
      for (int r = 0; r < 8; ++r)
        p2[(m0 + r) * 1024 + col] = acc[r] + bias;
    } else {
      int ct = t - 64;
      v8f acc = vzero8();
      for (int kt = 0; kt < 128; ++kt) {
        int tok = 4 * n + (kt >> 5);                 // reshape (8, 4*1024)
        if (tok > 31) tok = 31;                      // rows >=8 discarded
        const __bf16* arow = xm + tok * 1024 + (kt & 31) * 32;
        v16bf a = afrag(arow, 0, g);
        v16bf b = pp4[((size_t)(ct * 128 + kt)) * 32 + lane];
        acc = wmma_bf16(a, b, acc);
      }
      int col = ct * 16 + n;
      float bias = p4b[col];
#pragma unroll
      for (int r = 0; r < 8; ++r) {
        int mm = m0 + r;
        if (mm < 8) p4[mm * 1024 + col] = acc[r] + bias;
      }
    }
  }
  __syncthreads();

  float ms = miscf[0];
  for (int i = tid; i < 32768; i += NTHREADS) {
    int t = i >> 10, d = i & 1023;
    int c = chos[t];
    float val;
    if (c == 0) {
      val = xg[i];
    } else if (c == 1) {
      float m1 = ((float)t < ms * 0.5f) ? 1.f : 0.f;
      float s = (t < 16) ? p2[t * 1024 + d] : 0.f;
      val = s * m1 + femb[d];
    } else {
      float m2 = ((float)t < ms * 0.25f) ? 1.f : 0.f;
      float s = (t < 8) ? p4[t * 1024 + d] : 0.f;
      val = s * m2 + femb[1024 + d];
    }
    out[(size_t)seq * 32768 + i] = val;
  }
}

// ---------------------------------------------------------------------------
// Workspace layout (bytes)
// ---------------------------------------------------------------------------
static const size_t OFF_QKV = 0;                                  // 2*1024*3072*2
static const size_t OFF_WO  = OFF_QKV + (size_t)2 * 1024 * 3072 * 2;
static const size_t OFF_W1  = OFF_WO  + (size_t)2 * 1024 * 1024 * 2;
static const size_t OFF_W2  = OFF_W1  + (size_t)2 * 1024 * 4096 * 2;
static const size_t OFF_P2  = OFF_W2  + (size_t)2 * 4096 * 1024 * 2;
static const size_t OFF_P4  = OFF_P2  + (size_t)2048 * 1024 * 2;
static const size_t OFF_CH  = OFF_P4  + (size_t)4096 * 1024 * 2;

extern "C" void kernel_launch(void* const* d_in, const int* in_sizes, int n_in,
                              void* d_out, int out_size, void* d_ws, size_t ws_size,
                              hipStream_t stream) {
  const float* x     = (const float*)d_in[0];
  const float* amask = (const float*)d_in[1];
  const float* ln1w  = (const float*)d_in[2];
  const float* ln1b  = (const float*)d_in[3];
  const float* wqkv  = (const float*)d_in[4];
  const float* bqkv  = (const float*)d_in[5];
  const float* wo    = (const float*)d_in[6];
  const float* bo    = (const float*)d_in[7];
  const float* ln2w  = (const float*)d_in[8];
  const float* ln2b  = (const float*)d_in[9];
  const float* w1    = (const float*)d_in[10];
  const float* b1    = (const float*)d_in[11];
  const float* w2    = (const float*)d_in[12];
  const float* b2    = (const float*)d_in[13];
  const float* rfw   = (const float*)d_in[14];
  const float* rfb   = (const float*)d_in[15];
  const float* p2w   = (const float*)d_in[16];
  const float* p2b   = (const float*)d_in[17];
  const float* p4w   = (const float*)d_in[18];
  const float* p4b   = (const float*)d_in[19];
  const float* femb  = (const float*)d_in[20];

  char* ws = (char*)d_ws;
  __bf16* pk_qkv = (__bf16*)(ws + OFF_QKV);
  __bf16* pk_wo  = (__bf16*)(ws + OFF_WO);
  __bf16* pk_w1  = (__bf16*)(ws + OFF_W1);
  __bf16* pk_w2  = (__bf16*)(ws + OFF_W2);
  __bf16* pk_p2  = (__bf16*)(ws + OFF_P2);
  __bf16* pk_p4  = (__bf16*)(ws + OFF_P4);
  int*    chosen = (int*)(ws + OFF_CH);

  auto pack = [&](const float* W, __bf16* o, int K, int N) {
    long total = (long)K * N;
    int blocks = (int)((total + 255) / 256);
    pack_weight_kernel<<<blocks, 256, 0, stream>>>(W, o, K, N);
  };
  for (int l = 0; l < 2; ++l) {
    pack(wqkv + (size_t)l * 1024 * 3072, pk_qkv + (size_t)l * 1024 * 3072, 1024, 3072);
    pack(wo   + (size_t)l * 1024 * 1024, pk_wo  + (size_t)l * 1024 * 1024, 1024, 1024);
    pack(w1   + (size_t)l * 1024 * 4096, pk_w1  + (size_t)l * 1024 * 4096, 1024, 4096);
    pack(w2   + (size_t)l * 4096 * 1024, pk_w2  + (size_t)l * 4096 * 1024, 4096, 1024);
  }
  pack(p2w, pk_p2, 2048, 1024);
  pack(p4w, pk_p4, 4096, 1024);

  float* out_main  = (float*)d_out;
  float* masks_out = out_main + (size_t)32 * 1024 * 1024;
  float* comp_out  = masks_out + (size_t)32 * 1024;

  router_kernel<<<1024, NTHREADS, RTR_LDS, stream>>>(
      x, amask, ln1w, ln1b, bqkv, bo, ln2w, ln2b, b1, b2, rfw, rfb,
      pk_qkv, pk_wo, pk_w1, pk_w2, chosen, masks_out, comp_out);

  finalize_kernel<<<1024, NTHREADS, FIN_LDS, stream>>>(
      x, amask, pk_p2, pk_p4, p2b, p4b, femb, chosen, out_main);
}
